// DynSMHALayer_84928683311854
// MI455X (gfx1250) — matrix operand order repaired and checked by reading, not compiled, not run on hardware
//
#include <hip/hip_runtime.h>

// ---------------- problem constants ----------------
constexpr int Bn = 4, Tn = 2048, Cn = 2048, Dn = 128, En = 16;
constexpr int Nn = Bn * Tn;     // 8192 tokens

// ---------------- CDNA5 feature detection ----------------
#ifdef __has_builtin
#  if __has_builtin(__builtin_amdgcn_global_load_async_to_lds_b128)
#    define HAS_ASYNC_LDS 1
#  endif
#endif
#ifndef HAS_ASYNC_LDS
#  define HAS_ASYNC_LDS 0
#endif

// ---------------- WMMA plumbing (CDNA5 16x16x32 f16 -> f32) ----------------
typedef __attribute__((ext_vector_type(16))) _Float16 v16h;
typedef __attribute__((ext_vector_type(8)))  float    v8f;

union V16H { v16h v; unsigned int u[8]; };

__device__ __forceinline__ v8f wmma16(v16h a, v16h b, v8f c) {
    return __builtin_amdgcn_wmma_f32_16x16x32_f16(
        false, a, false, b, (short)0, c, /*reuse_a=*/false, /*reuse_b=*/false);
}
// same, but hint HW to cache the A operand for the next (identical) WMMA
__device__ __forceinline__ v8f wmma16_ra(v16h a, v16h b, v8f c) {
    return __builtin_amdgcn_wmma_f32_16x16x32_f16(
        false, a, false, b, (short)0, c, /*reuse_a=*/true, /*reuse_b=*/false);
}

// A-matrix 16x32 f16: lane L holds row M=L%16; VGPR g holds K pair:
// K = 16*(g/4) + 8*(L/16) + 2*(g%4)
__device__ __forceinline__ int a_koff(int g, int half) {
    return ((g >> 2) << 4) + (half << 3) + ((g & 3) << 1);
}
// B-matrix 32x16 f16: lane L holds col N=L%16; VGPR g holds K pair:
__device__ __forceinline__ int b_koff(int g, int half) {
    return ((g >> 2) << 4) + (half << 3) + ((g & 3) << 1);
}

__device__ __forceinline__ v8f v8f_zero() {
    v8f z = {0.f, 0.f, 0.f, 0.f, 0.f, 0.f, 0.f, 0.f};
    return z;
}

// 16-byte global->LDS stage (async path; builtin takes int4 AS1* / int4 AS3*)
typedef int v4i __attribute__((vector_size(16)));
__device__ __forceinline__ void stage16(const _Float16* gsrc, _Float16* ldst) {
#if HAS_ASYNC_LDS
    __builtin_amdgcn_global_load_async_to_lds_b128(
        (__attribute__((address_space(1))) v4i*)(v4i*)(void*)gsrc,
        (__attribute__((address_space(3))) v4i*)(v4i*)(void*)ldst,
        /*offset=*/0, /*cpol=*/0);
#else
    *(uint4*)ldst = *(const uint4*)gsrc;
#endif
}
__device__ __forceinline__ void wait_async_le1() {
#if HAS_ASYNC_LDS
    asm volatile("s_wait_asynccnt 0x1" ::: "memory");
#endif
}
__device__ __forceinline__ void wait_async_le0() {
#if HAS_ASYNC_LDS
    asm volatile("s_wait_asynccnt 0x0" ::: "memory");
#endif
}

// ---------------- conversion kernels ----------------
__global__ __launch_bounds__(256) void cvt_f16(const float* __restrict__ in,
                                               _Float16* __restrict__ out, int n) {
    int i = blockIdx.x * 256 + threadIdx.x;
    if (i < n) out[i] = (_Float16)in[i];
}

// [E, R, S] fp32 -> [E, S, R] f16
__global__ __launch_bounds__(256) void cvt_f16_T(const float* __restrict__ in,
                                                 _Float16* __restrict__ out,
                                                 int R, int S) {
    int i = blockIdx.x * 256 + threadIdx.x;
    int total = En * R * S;
    if (i >= total) return;
    int e = i / (R * S);
    int rem = i - e * (R * S);
    int r = rem / S;
    int s = rem - r * S;
    out[((size_t)e * S + s) * R + r] = (_Float16)in[i];
}

// ---------------- gating ----------------
// out[0..15] = column norms of sim_matrix, out[16..31] = sigmoid(gates)
__global__ __launch_bounds__(256) void prep_gating(const float* __restrict__ sim,
                                                   const float* __restrict__ gates,
                                                   float* __restrict__ out) {
    __shared__ float red[256];
    int t = threadIdx.x, e = t & 15, s = t >> 4;    // 16 slices x 16 experts
    float acc = 0.f;
    for (int i = 0; i < Cn / 16; ++i) {
        float v = sim[(s * (Cn / 16) + i) * En + e];
        acc += v * v;
    }
    red[t] = acc;
    __syncthreads();
    if (t < 16) {
        float sum = 0.f;
        for (int s2 = 0; s2 < 16; ++s2) sum += red[s2 * 16 + t];
        out[t] = fmaxf(sqrtf(sum), 1e-12f);
        out[16 + t] = 1.f / (1.f + __expf(-gates[t]));
    }
}

// one block per token: routing weights rw[token, e]
__global__ __launch_bounds__(256) void gating_kernel(const float* __restrict__ hs,
                                                     const float* __restrict__ sim,
                                                     const float* __restrict__ ss,
                                                     const int* __restrict__ minE_p,
                                                     float* __restrict__ rw) {
    __shared__ float dred[256];
    __shared__ float xred[16];
    const int tok = blockIdx.x;
    const int t = threadIdx.x, e = t & 15, s = t >> 4;
    const float* x = hs + (size_t)tok * Cn;
    float dacc = 0.f, xacc = 0.f;
    for (int i = 0; i < Cn / 16; ++i) {
        int c = s * (Cn / 16) + i;
        float xv = x[c];
        dacc += xv * sim[c * En + e];
        if (e == 0) xacc += xv * xv;
    }
    dred[t] = dacc;
    if (e == 0) xred[s] = xacc;
    __syncthreads();
    if (t == 0) {
        float xs = 0.f;
        for (int s2 = 0; s2 < 16; ++s2) xs += xred[s2];
        float xn = fmaxf(sqrtf(xs), 1e-12f);
        float logits[En], gated[En];
        bool act[En];
        int nact = 0;
        for (int e2 = 0; e2 < En; ++e2) {
            float d = 0.f;
            for (int s2 = 0; s2 < 16; ++s2) d += dred[s2 * 16 + e2];
            float lg = d / (xn * ss[e2]) - ss[16 + e2];
            logits[e2] = lg;
            gated[e2] = lg > 0.f ? lg : 0.f;
            act[e2] = lg > 0.f;
            if (act[e2]) ++nact;
        }
        if (nact == 0) {
            int kmin = *minE_p;
            bool chosen[En];
            for (int e2 = 0; e2 < En; ++e2) chosen[e2] = false;
            for (int k2 = 0; k2 < kmin; ++k2) {
                int bi = 0;
                float bv = -3.0e38f;
                for (int e2 = 0; e2 < En; ++e2)
                    if (!chosen[e2] && logits[e2] > bv) { bv = logits[e2]; bi = e2; }
                chosen[bi] = true;
            }
            for (int e2 = 0; e2 < En; ++e2) act[e2] = chosen[e2];
        }
        float mx = -3.0e38f;
        for (int e2 = 0; e2 < En; ++e2)
            if (act[e2]) mx = fmaxf(mx, gated[e2]);
        float sum = 0.f, w[En];
        for (int e2 = 0; e2 < En; ++e2) {
            float v = act[e2] ? __expf(gated[e2] - mx) : 0.f;
            w[e2] = v; sum += v;
        }
        float inv = 1.f / sum;
        for (int e2 = 0; e2 < En; ++e2) rw[(size_t)tok * En + e2] = w[e2] * inv;
    }
}

// ---------------- expert-weighted projection: out = sum_e rw_e * (X @ W_e) ----------------
// X: [N, C] f16; WT: [E, D, C] f16 (K contiguous); OUT f16.
// LDS-staged X tile (async double-buffered); j-outer/e-inner WMMA chains share the
// same A operand across 16 experts -> reuse_a hint; 16 accumulators live in VGPRs.
// storeT==0 -> OUT[token*D + d];  storeT==1 -> OUT[(b*D + d)*T + t]  (V transposed)
__global__ __launch_bounds__(256) void expert_proj(const _Float16* __restrict__ X,
                                                   const _Float16* __restrict__ WT,
                                                   const float* __restrict__ RW,
                                                   _Float16* __restrict__ OUT,
                                                   int storeT) {
    constexpr int CK = 128;                       // K chunk (channels per stage)
    constexpr int NCHUNK = Cn / CK;               // 16
    __shared__ __align__(16) _Float16 atile[2][16 * CK];   // 2 x 4KB

    const int lane = threadIdx.x & 31;
    const int wave = threadIdx.x >> 5;
    const int half = lane >> 4;
    const int ln = lane & 15;
    const int tok0 = blockIdx.x * 16;
    const int col0 = wave * 16;                   // 8 waves cover D=128

    // cooperative copy role: 256 threads x 16B = 16 rows x 128 halves
    const int crow = threadIdx.x >> 4;            // token row 0..15
    const int cseg = (threadIdx.x & 15) << 3;     // half offset 0,8,..,120
    const _Float16* gsrc = X + (size_t)(tok0 + crow) * Cn + cseg;
    _Float16* l0 = &atile[0][crow * CK + cseg];
    _Float16* l1 = &atile[1][crow * CK + cseg];

    v8f acc[En];
#pragma unroll
    for (int e = 0; e < En; ++e) acc[e] = v8f_zero();

    const _Float16* wcol = WT + (size_t)(col0 + ln) * Cn;   // B col = lane%16

    stage16(gsrc, l0);                             // prologue: chunk 0

    for (int k = 0; k < NCHUNK; ++k) {
        const int cur = k & 1;
        if (k + 1 < NCHUNK) {
            stage16(gsrc + (k + 1) * CK, cur ? l0 : l1);
            wait_async_le1();
        } else {
            wait_async_le0();
        }
        __syncthreads();                           // tile[cur] visible to all waves

        // prefetch next K chunk of every expert's weights into near cache
#pragma unroll
        for (int e = 0; e < En; ++e)
            __builtin_prefetch(wcol + (size_t)e * Dn * Cn + k * CK + CK, 0, 3);

        // A chunk -> registers (read once, reused by all 16 experts)
        V16H aR[4];
        const _Float16* lrow = &atile[cur][ln * CK];
#pragma unroll
        for (int j = 0; j < 4; ++j)
#pragma unroll
            for (int g = 0; g < 8; ++g)
                aR[j].u[g] = *(const unsigned int*)(lrow + j * 32 + a_koff(g, half));

        // j-outer: 16 consecutive WMMAs share A => reuse_a on all but the last
#pragma unroll
        for (int j = 0; j < 4; ++j) {
#pragma unroll
            for (int e = 0; e < En; ++e) {
                const _Float16* wb = wcol + (size_t)e * Dn * Cn + k * CK + j * 32;
                V16H b;
#pragma unroll
                for (int g = 0; g < 8; ++g)
                    b.u[g] = *(const unsigned int*)(wb + b_koff(g, half));
                acc[e] = (e < En - 1) ? wmma16_ra(aR[j].v, b.v, acc[e])
                                      : wmma16(aR[j].v, b.v, acc[e]);
            }
        }
        __syncthreads();                           // all done reading tile[cur]
    }

    // epilogue: routing-weighted sum over experts
    float fin[8];
#pragma unroll
    for (int r = 0; r < 8; ++r) fin[r] = 0.f;
#pragma unroll
    for (int e = 0; e < En; ++e)
#pragma unroll
        for (int r = 0; r < 8; ++r) {
            int tok = tok0 + r + (half << 3);
            fin[r] += RW[(size_t)tok * En + e] * acc[e][r];
        }
#pragma unroll
    for (int r = 0; r < 8; ++r) {
        int tok = tok0 + r + (half << 3);
        int d = col0 + ln;
        if (!storeT) {
            OUT[(size_t)tok * Dn + d] = (_Float16)fin[r];
        } else {
            int b_ = tok / Tn, t_ = tok - b_ * Tn;
            OUT[((size_t)b_ * Dn + d) * Tn + t_] = (_Float16)fin[r];
        }
    }
}

// ---------------- flash attention (causal), one q-tile (16 rows) per wave ----------------
// Q,K: [B*T, D] f16; VT: [B, D, T] f16; O: [B*T, D] f16
__global__ __launch_bounds__(128) void attention(const _Float16* __restrict__ Q,
                                                 const _Float16* __restrict__ K,
                                                 const _Float16* __restrict__ VT,
                                                 _Float16* __restrict__ O) {
    __shared__ __align__(16) _Float16 ldsP[4][16 * 32];   // per-wave P tile
    const int wave = threadIdx.x >> 5;
    const int lane = threadIdx.x & 31;
    const int half = lane >> 4;
    const int ln = lane & 15;
    const int b = blockIdx.y;
    const int q0 = (blockIdx.x * 4 + wave) * 16;
    const float scale = 0.08838834764831845f;             // 1/sqrt(128)

    // preload Q in A layout: 4 chunks over D
    V16H qa[4];
    const _Float16* qbase = Q + (size_t)(b * Tn + q0 + ln) * Dn;
#pragma unroll
    for (int j = 0; j < 4; ++j)
#pragma unroll
        for (int g = 0; g < 8; ++g)
            qa[j].u[g] = *(const unsigned int*)(qbase + j * 32 + a_koff(g, half));

    float mrow[8], lrow[8];
    v8f oacc[8];
#pragma unroll
    for (int r = 0; r < 8; ++r) { mrow[r] = -3.0e38f; lrow[r] = 0.f; }
#pragma unroll
    for (int ct = 0; ct < 8; ++ct) oacc[ct] = v8f_zero();

    _Float16* lp = ldsP[wave];

    for (int s0 = 0; s0 <= q0 + 15; s0 += 32) {
        // S = Q @ K^T  (two 16x16 tiles covering kv s0..s0+31)
        v8f s[2];
#pragma unroll
        for (int t2 = 0; t2 < 2; ++t2) {
            v8f acc = v8f_zero();
            const _Float16* kbase = K + (size_t)(b * Tn + s0 + t2 * 16 + ln) * Dn;
#pragma unroll
            for (int j = 0; j < 4; ++j) {
                V16H kb;
#pragma unroll
                for (int g = 0; g < 8; ++g)
                    kb.u[g] = *(const unsigned int*)(kbase + j * 32 + b_koff(g, half));
                acc = wmma16(qa[j].v, kb.v, acc);
            }
            s[t2] = acc;
        }
        // scale + causal mask
#pragma unroll
        for (int r = 0; r < 8; ++r) {
            int qi = q0 + r + (half << 3);
#pragma unroll
            for (int t2 = 0; t2 < 2; ++t2) {
                int si = s0 + t2 * 16 + ln;
                float v = s[t2][r] * scale;
                if (si > qi) v = -3.0e38f;
                s[t2][r] = v;
            }
        }
        // online softmax: row max / exp / row sum (row spans 16 lanes of a half)
        float al[8];
#pragma unroll
        for (int r = 0; r < 8; ++r) {
            float mx = fmaxf(s[0][r], s[1][r]);
            for (int off = 1; off < 16; off <<= 1)
                mx = fmaxf(mx, __shfl_xor(mx, off, 32));
            float mnew = fmaxf(mrow[r], mx);
            al[r] = __expf(mrow[r] - mnew);
            mrow[r] = mnew;
            float p0 = __expf(s[0][r] - mnew);
            float p1 = __expf(s[1][r] - mnew);
            s[0][r] = p0; s[1][r] = p1;
            float sum = p0 + p1;
            for (int off = 1; off < 16; off <<= 1)
                sum += __shfl_xor(sum, off, 32);
            lrow[r] = lrow[r] * al[r] + sum;
        }
#pragma unroll
        for (int ct = 0; ct < 8; ++ct)
#pragma unroll
            for (int r = 0; r < 8; ++r) oacc[ct][r] *= al[r];

        // C-layout -> A-layout transpose of P through LDS (same-wave DS ops are in-order)
#pragma unroll
        for (int r = 0; r < 8; ++r) {
            int m = r + (half << 3);
#pragma unroll
            for (int t2 = 0; t2 < 2; ++t2)
                lp[m * 32 + t2 * 16 + ln] = (_Float16)s[t2][r];
        }
        V16H pa;
#pragma unroll
        for (int g = 0; g < 8; ++g)
            pa.u[g] = *(const unsigned int*)(lp + ln * 32 + a_koff(g, half));

        // O += P @ V : 8 consecutive WMMAs share A (pa) => reuse_a on all but last
#pragma unroll
        for (int ct = 0; ct < 8; ++ct) {
            V16H vb;
            const _Float16* vbase = VT + ((size_t)b * Dn + ct * 16 + ln) * Tn + s0;
#pragma unroll
            for (int g = 0; g < 8; ++g)
                vb.u[g] = *(const unsigned int*)(vbase + b_koff(g, half));
            oacc[ct] = (ct < 7) ? wmma16_ra(pa.v, vb.v, oacc[ct])
                                : wmma16(pa.v, vb.v, oacc[ct]);
        }
    }
    // epilogue: normalize and store f16
#pragma unroll
    for (int ct = 0; ct < 8; ++ct) {
#pragma unroll
        for (int r = 0; r < 8; ++r) {
            int tok = q0 + r + (half << 3);
            int d = ct * 16 + ln;
            float v = oacc[ct][r] / lrow[r];
            O[(size_t)(b * Tn + tok) * Dn + d] = (_Float16)v;
        }
    }
}

// ---------------- expert-weighted output projection ----------------
// O: [N, D] f16; WT: [E, C, D] f16 (o_proj transposed, K=d contiguous); OUT: [N, C] fp32
__global__ __launch_bounds__(256) void out_proj(const _Float16* __restrict__ O,
                                                const _Float16* __restrict__ WT,
                                                const float* __restrict__ RW,
                                                float* __restrict__ OUT) {
    const int wave = threadIdx.x >> 5;
    const int lane = threadIdx.x & 31;
    const int half = lane >> 4;
    const int ln = lane & 15;
    const int tok0 = blockIdx.x * 16;
    const int c0 = blockIdx.y * 128 + wave * 16;

    // preload A (o tile), 4 chunks over D — shared by all experts
    V16H a[4];
    const _Float16* obase = O + (size_t)(tok0 + ln) * Dn;
#pragma unroll
    for (int j = 0; j < 4; ++j)
#pragma unroll
        for (int g = 0; g < 8; ++g)
            a[j].u[g] = *(const unsigned int*)(obase + j * 32 + a_koff(g, half));

    v8f acc[En];
#pragma unroll
    for (int e = 0; e < En; ++e) acc[e] = v8f_zero();

    const _Float16* wcol = WT + (size_t)(c0 + ln) * Dn;

    // j-outer: 16 consecutive WMMAs share A => reuse_a on all but the last
#pragma unroll
    for (int j = 0; j < 4; ++j) {
#pragma unroll
        for (int e = 0; e < En; ++e) {
            const _Float16* wb = wcol + (size_t)e * Cn * Dn + j * 32;
            V16H bmat;
#pragma unroll
            for (int g = 0; g < 8; ++g)
                bmat.u[g] = *(const unsigned int*)(wb + b_koff(g, half));
            acc[e] = (e < En - 1) ? wmma16_ra(a[j].v, bmat.v, acc[e])
                                  : wmma16(a[j].v, bmat.v, acc[e]);
        }
    }

    float fin[8];
#pragma unroll
    for (int r = 0; r < 8; ++r) fin[r] = 0.f;
#pragma unroll
    for (int e = 0; e < En; ++e)
#pragma unroll
        for (int r = 0; r < 8; ++r) {
            int tok = tok0 + r + (half << 3);
            fin[r] += RW[(size_t)tok * En + e] * acc[e][r];
        }
#pragma unroll
    for (int r = 0; r < 8; ++r) {
        int tok = tok0 + r + (half << 3);
        OUT[(size_t)tok * Cn + c0 + ln] = fin[r];
    }
}

// ---------------- launch ----------------
extern "C" void kernel_launch(void* const* d_in, const int* in_sizes, int n_in,
                              void* d_out, int out_size, void* d_ws, size_t ws_size,
                              hipStream_t stream) {
    const float* hs    = (const float*)d_in[0];   // [B,T,C]
    const float* sim   = (const float*)d_in[1];   // [C,E]
    const float* gates = (const float*)d_in[2];   // [E]
    const float* qp    = (const float*)d_in[3];   // [E,C,D]
    const float* kp    = (const float*)d_in[4];
    const float* vp    = (const float*)d_in[5];
    const float* op    = (const float*)d_in[6];   // [E,D,C]
    const int*   minE  = (const int*)d_in[7];
    float* out = (float*)d_out;

    char* w = (char*)d_ws;
    auto alloc = [&](size_t bytes) -> char* {
        char* p = w;
        w += (bytes + 255) & ~(size_t)255;
        return p;
    };
    _Float16* xh   = (_Float16*)alloc((size_t)Nn * Cn * 2);        // hidden f16
    _Float16* qT   = (_Float16*)alloc((size_t)En * Dn * Cn * 2);   // [E,D,C]
    _Float16* kT   = (_Float16*)alloc((size_t)En * Dn * Cn * 2);
    _Float16* vT   = (_Float16*)alloc((size_t)En * Dn * Cn * 2);
    _Float16* oT   = (_Float16*)alloc((size_t)En * Cn * Dn * 2);   // [E,C,D]
    float*    rw   = (float*)   alloc((size_t)Nn * En * 4);
    float*    ss   = (float*)   alloc(32 * 4);
    _Float16* qb   = (_Float16*)alloc((size_t)Nn * Dn * 2);        // [N,D]
    _Float16* kb   = (_Float16*)alloc((size_t)Nn * Dn * 2);
    _Float16* vtb  = (_Float16*)alloc((size_t)Nn * Dn * 2);        // [B,D,T]
    _Float16* ob   = (_Float16*)alloc((size_t)Nn * Dn * 2);        // [N,D]

    const int nhid = Nn * Cn;
    cvt_f16<<<(nhid + 255) / 256, 256, 0, stream>>>(hs, xh, nhid);
    const int nw = En * Cn * Dn;
    cvt_f16_T<<<(nw + 255) / 256, 256, 0, stream>>>(qp, qT, Cn, Dn);
    cvt_f16_T<<<(nw + 255) / 256, 256, 0, stream>>>(kp, kT, Cn, Dn);
    cvt_f16_T<<<(nw + 255) / 256, 256, 0, stream>>>(vp, vT, Cn, Dn);
    cvt_f16_T<<<(nw + 255) / 256, 256, 0, stream>>>(op, oT, Dn, Cn);

    prep_gating<<<1, 256, 0, stream>>>(sim, gates, ss);
    gating_kernel<<<Nn, 256, 0, stream>>>(hs, sim, ss, minE, rw);

    expert_proj<<<Nn / 16, 256, 0, stream>>>(xh, qT, rw, qb, 0);
    expert_proj<<<Nn / 16, 256, 0, stream>>>(xh, kT, rw, kb, 0);
    expert_proj<<<Nn / 16, 256, 0, stream>>>(xh, vT, rw, vtb, 1);

    attention<<<dim3(Tn / 64, Bn), 128, 0, stream>>>(qb, kb, vtb, ob);

    out_proj<<<dim3(Nn / 16, Cn / 128), 256, 0, stream>>>(ob, oT, rw, out);

    (void)in_sizes; (void)n_in; (void)out_size; (void)ws_size;
}
